// CMDNetBinaryLegacy_73478300500640
// MI455X (gfx1250) — matrix operand order repaired and checked by reading, not compiled, not run on hardware
//
#include <hip/hip_runtime.h>

typedef __attribute__((ext_vector_type(16))) _Float16 v16h;
typedef __attribute__((ext_vector_type(8)))  _Float16 v8h;
typedef __attribute__((ext_vector_type(8)))  float    v8f;
typedef __attribute__((ext_vector_type(4)))  float    v4f;

#define WARPS_PER_BLOCK 4
#define N_DIM 64
#define M_DIM 32
#define IT_STEPS 10
// halfs per Ht^T row in LDS: >=64, multiple of 8 (16B) so every fragment
// sub-load is a 16B-aligned ds_load_b128; 72 also breaks the worst bank pattern.
#define LDS_STRIDE 72

__device__ __forceinline__ v16h cat8h(v8h lo, v8h hi) {
    return __builtin_shufflevector(lo, hi, 0,1,2,3,4,5,6,7,8,9,10,11,12,13,14,15);
}

__global__ __launch_bounds__(WARPS_PER_BLOCK * 32)
void cmdnet_binary_kernel(const float* __restrict__ yt,
                          const float* __restrict__ Ht,
                          const float* __restrict__ sigmat0,
                          const float* __restrict__ taui,
                          const float* __restrict__ delta,
                          const float* __restrict__ alphat,
                          const float* __restrict__ mvec,
                          float* __restrict__ out,
                          int B)
{
    __shared__ __align__(16) _Float16 sHtT[WARPS_PER_BLOCK][M_DIM * LDS_STRIDE]; // Ht^T, f16
    __shared__ float sHH[WARPS_PER_BLOCK][M_DIM * M_DIM];                        // Gram matrix
    __shared__ float sYt[WARPS_PER_BLOCK][N_DIM];                                // yt staging
    __shared__ float sXt[WARPS_PER_BLOCK][M_DIM];                                // xt broadcast

    const int lane = threadIdx.x & 31;
    const int warp = threadIdx.x >> 5;
    const int b    = blockIdx.x * WARPS_PER_BLOCK + warp;
    if (b >= B) return;                 // wave-uniform: EXEC stays all-ones

    _Float16* __restrict__ htT = sHtT[warp];
    float*    __restrict__ hh  = sHH[warp];
    float*    __restrict__ yts = sYt[warp];
    float*    __restrict__ xts = sXt[warp];

    const float* HtB = Ht + (size_t)b * (N_DIM * M_DIM);

    // ---- stage yt (f32) ----
    yts[lane]      = yt[(size_t)b * N_DIM + lane];
    yts[lane + 32] = yt[(size_t)b * N_DIM + lane + 32];

    // ---- stream Ht once, coalesced b128, transpose+convert into LDS f16 [m][n] ----
    #pragma unroll
    for (int i = 0; i < 16; ++i) {
        const int off = i * 128 + lane * 4;            // flat float index; wave covers 512B/iter
        v4f v = *(const v4f*)(HtB + off);
        const int n = off >> 5;                        // row (n) of Ht
        const int m = off & 31;                        // first col (m)
        htT[(m + 0) * LDS_STRIDE + n] = (_Float16)v[0];
        htT[(m + 1) * LDS_STRIDE + n] = (_Float16)v[1];
        htT[(m + 2) * LDS_STRIDE + n] = (_Float16)v[2];
        htT[(m + 3) * LDS_STRIDE + n] = (_Float16)v[3];
    }
    __syncthreads();

    const int h   = lane >> 4;   // lane group (0/1)
    const int l16 = lane & 15;

    // ---- HH = Ht^T x Ht via 8x v_wmma_f32_16x16x32_f16 (2x2 tiles, K = 2x32) ----
    v8f acc00 = {}, acc01 = {}, acc10 = {}, acc11 = {};
    #pragma unroll
    for (int kc = 0; kc < 2; ++kc) {
        // A fragments (ISA 16-bit A 16x32 layout): elem e -> K = 16*(e/8) + 8*h + e%8
        const _Float16* pa0 = &htT[(l16)      * LDS_STRIDE + kc * 32 + 8 * h];
        const _Float16* pa1 = &htT[(16 + l16) * LDS_STRIDE + kc * 32 + 8 * h];
        v16h a0 = cat8h(*(const v8h*)pa0, *(const v8h*)(pa0 + 16));
        v16h a1 = cat8h(*(const v8h*)pa1, *(const v8h*)(pa1 + 16));
        // B fragments (same as A): elem e -> K = 16*(e/8) + 8*h + e%8
        const _Float16* pb0 = &htT[(l16)      * LDS_STRIDE + kc * 32 + 8 * h];
        const _Float16* pb1 = &htT[(16 + l16) * LDS_STRIDE + kc * 32 + 8 * h];
        v16h b0 = cat8h(*(const v8h*)pb0, *(const v8h*)(pb0 + 16));
        v16h b1 = cat8h(*(const v8h*)pb1, *(const v8h*)(pb1 + 16));

        acc00 = __builtin_amdgcn_wmma_f32_16x16x32_f16(false, a0, false, b0, (short)0, acc00, false, false);
        acc01 = __builtin_amdgcn_wmma_f32_16x16x32_f16(false, a0, false, b1, (short)0, acc01, false, false);
        acc10 = __builtin_amdgcn_wmma_f32_16x16x32_f16(false, a1, false, b0, (short)0, acc10, false, false);
        acc11 = __builtin_amdgcn_wmma_f32_16x16x32_f16(false, a1, false, b1, (short)0, acc11, false, false);
    }

    // ---- scatter accumulators (C/D layout: VGPR r, lane -> M=r+8h, N=l16) into HH ----
    #pragma unroll
    for (int r = 0; r < 8; ++r) {
        const int row = r + 8 * h;
        hh[(row)      * M_DIM + l16]      = acc00[r];
        hh[(row)      * M_DIM + 16 + l16] = acc01[r];
        hh[(16 + row) * M_DIM + l16]      = acc10[r];
        hh[(16 + row) * M_DIM + 16 + l16] = acc11[r];
    }

    // ---- yH[m] = sum_n yt[n] * Ht[n][m]   (lane == m; contiguous f16 row in LDS) ----
    float yH = 0.0f;
    #pragma unroll
    for (int n = 0; n < N_DIM; ++n)
        yH += yts[n] * (float)htT[lane * LDS_STRIDE + n];

    __syncthreads();   // HH visible across lanes

    // ---- iteration state (lane == m) ----
    const float sigv = sigmat0[b];
    const float sig2 = sigv * sigv;
    const float bias = logf(1.0f / alphat[lane] - 1.0f);
    float s        = 0.0f;
    float taui_abs = fabsf(taui[0]);
    float xt       = tanhf(bias * 0.5f * taui_abs);

    #pragma unroll
    for (int it = 0; it < IT_STEPS; ++it) {
        xts[lane] = xt;
        __syncthreads();
        float xHH = 0.0f;
        #pragma unroll
        for (int mm = 0; mm < M_DIM; ++mm)
            xHH += xts[mm] * hh[mm * M_DIM + lane];   // conflict-free column read
        __syncthreads();

        const float d      = delta[it];
        const float grad_x = 0.5f * taui_abs * (1.0f - xt * xt);
        const float grad_L = sig2 * tanhf(s * 0.5f) + grad_x * (xHH - yH);
        s        = s - d * grad_L;
        taui_abs = fabsf(taui[it + 1]);
        xt       = tanhf((bias + s) * 0.5f * taui_abs);
    }

    // ---- outputs: ft (B,32,2) then xt (B,32), concatenated flat ----
    const float m0 = mvec[0];
    const float m1 = mvec[1];
    float2 ft;
    ft.x = (1.0f + m0 * xt) * 0.5f;
    ft.y = (1.0f + m1 * xt) * 0.5f;
    *(float2*)(out + (size_t)b * (M_DIM * 2) + lane * 2) = ft;     // coalesced b64
    out[(size_t)B * (M_DIM * 2) + (size_t)b * M_DIM + lane] = xt;
}

extern "C" void kernel_launch(void* const* d_in, const int* in_sizes, int n_in,
                              void* d_out, int out_size, void* d_ws, size_t ws_size,
                              hipStream_t stream)
{
    const float* yt      = (const float*)d_in[0];
    const float* Ht      = (const float*)d_in[1];
    const float* sigmat0 = (const float*)d_in[2];
    const float* taui    = (const float*)d_in[3];
    const float* delta   = (const float*)d_in[4];
    const float* alphat  = (const float*)d_in[5];
    const float* mvec    = (const float*)d_in[6];
    float* out = (float*)d_out;

    const int B = in_sizes[2];                       // sigmat0 has one entry per batch
    const int blocks = (B + WARPS_PER_BLOCK - 1) / WARPS_PER_BLOCK;

    hipLaunchKernelGGL(cmdnet_binary_kernel, dim3(blocks), dim3(WARPS_PER_BLOCK * 32), 0, stream,
                       yt, Ht, sigmat0, taui, delta, alphat, mvec, out, B);
}
